// MultiHeadAttention_65438121722131
// MI455X (gfx1250) — matrix operand order repaired and checked by reading, not compiled, not run on hardware
//
#include <hip/hip_runtime.h>

#define D_MODEL   768
#define NUM_HEADS 12
#define D_K       64
#define SEQ       2048
#define BATCH     2

typedef __attribute__((ext_vector_type(16))) __bf16 v16bf;
typedef __attribute__((ext_vector_type(8)))  __bf16 bf16x8;   // one b128 load
typedef __attribute__((ext_vector_type(8)))  float  v8f;
typedef __attribute__((ext_vector_type(4)))  float  f32x4;    // one b128 load

// ---------------------------------------------------------------------------
// WMMA: D = A(16x32 bf16) * B(32x16 bf16) + C(16x16 f32), wave32.
// ---------------------------------------------------------------------------
__device__ __forceinline__ v8f wmma_bf16(v16bf a, v16bf b, v8f c) {
  return __builtin_amdgcn_wmma_f32_16x16x32_bf16(
      false, a, false, b, (short)0, c, false, false);
}

__device__ __forceinline__ v16bf cat8(bf16x8 lo, bf16x8 hi) {
  return __builtin_shufflevector(lo, hi, 0, 1, 2, 3, 4, 5, 6, 7,
                                 8, 9, 10, 11, 12, 13, 14, 15);
}

// A fragment, bf16 row-major [.,lda]. ISA 7.12.2: lane L row m=L%16; VGPR j
// holds K=(j/4)*16+(L/16)*8+(j%4)*2 -> per lane two contiguous 16B chunks.
__device__ __forceinline__ v16bf load_a_bf16(const __bf16* __restrict__ A, int lda,
                                             int m0, int kc, int lane) {
  const __bf16* row =
      A + (size_t)(m0 + (lane & 15)) * lda + kc + ((lane >> 4) << 3);
  return cat8(*(const bf16x8*)(row), *(const bf16x8*)(row + 16));
}

// Same A fragment from f32 memory: 4 x b128 + packed cvt.
__device__ __forceinline__ v16bf load_a_f32(const float* __restrict__ A, int lda,
                                            int m0, int kc, int lane) {
  const float* row =
      A + (size_t)(m0 + (lane & 15)) * lda + kc + ((lane >> 4) << 3);
  f32x4 x0 = *(const f32x4*)(row);
  f32x4 x1 = *(const f32x4*)(row + 4);
  f32x4 x2 = *(const f32x4*)(row + 16);
  f32x4 x3 = *(const f32x4*)(row + 20);
  v16bf a;
#pragma unroll
  for (int i = 0; i < 4; ++i) {
    a[i]      = (__bf16)x0[i];
    a[4 + i]  = (__bf16)x1[i];
    a[8 + i]  = (__bf16)x2[i];
    a[12 + i] = (__bf16)x3[i];
  }
  return a;
}

// B fragment from Bt[n][k] bf16 row-major (computes *Bt^T): lane L col n=L%16;
// same K layout as A: two 16B chunks at k=(L/16)*8 and k=16+(L/16)*8.
__device__ __forceinline__ v16bf load_bT_bf16(const __bf16* __restrict__ Bt, int ldt,
                                              int kc, int n0, int lane) {
  const __bf16* row =
      Bt + (size_t)(n0 + (lane & 15)) * ldt + kc + ((lane >> 4) << 3);
  return cat8(*(const bf16x8*)(row), *(const bf16x8*)(row + 16));
}

// ---------------------------------------------------------------------------
// Weight transpose+convert: Wt[n][k] bf16 <- W[k][n] f32 (768x768, tiny).
// ---------------------------------------------------------------------------
__global__ void mha_transpose_convert(const float* __restrict__ W,
                                      __bf16* __restrict__ Wt) {
  const int i = blockIdx.x * 256 + threadIdx.x;     // coalesced read along n
  const int k = i / D_MODEL, n = i % D_MODEL;
  Wt[(size_t)n * D_MODEL + k] = (__bf16)W[i];
}

// ---------------------------------------------------------------------------
// GEMM: Y[M,N] = X[M,K] @ Wt^T (+bias). One wave = 32x64 tile (8 WMMAs/k-step).
// MODE 0: f32 store + bias (output projection -> d_out)
// MODE 1: bf16 row-major store (Q,K projections)
// MODE 2: bf16 transposed head store Vt[b,h,d,s] (V projection)
// ABF: A operand is bf16 (output projection reads bf16 attn_out).
// ---------------------------------------------------------------------------
template <int MODE, bool ABF>
__global__ void mha_gemm(const float* __restrict__ Xf, const __bf16* __restrict__ Xb,
                         const __bf16* __restrict__ Wt, const float* __restrict__ bias,
                         float* __restrict__ Yf, __bf16* __restrict__ Yb,
                         int M, int N, int K) {
  const int lane = threadIdx.x & 31;
  const int id   = blockIdx.x * (blockDim.x >> 5) + (threadIdx.x >> 5);
  const int nStrips = N >> 6;
  if (id >= (M >> 5) * nStrips) return;             // wave-uniform
  const int m0 = (id / nStrips) * 32;
  const int n0 = (id % nStrips) * 64;

  v8f acc[2][4] = {};
  for (int kc = 0; kc < K; kc += 32) {
    if (kc + 32 < K) {                              // gfx1250 global_prefetch_b8
      if (ABF) __builtin_prefetch(Xb + (size_t)(m0 + lane) * K + kc + 32, 0, 1);
      else     __builtin_prefetch(Xf + (size_t)(m0 + lane) * K + kc + 32, 0, 1);
    }
    v16bf a0 = ABF ? load_a_bf16(Xb, K, m0,      kc, lane)
                   : load_a_f32 (Xf, K, m0,      kc, lane);
    v16bf a1 = ABF ? load_a_bf16(Xb, K, m0 + 16, kc, lane)
                   : load_a_f32 (Xf, K, m0 + 16, kc, lane);
#pragma unroll
    for (int t = 0; t < 4; ++t) {
      v16bf bt = load_bT_bf16(Wt, K, kc, n0 + t * 16, lane);
      acc[0][t] = wmma_bf16(a0, bt, acc[0][t]);
      acc[1][t] = wmma_bf16(a1, bt, acc[1][t]);
    }
  }

#pragma unroll
  for (int hlf = 0; hlf < 2; ++hlf) {
    const int mb = m0 + hlf * 16 + ((lane >> 4) << 3);
#pragma unroll
    for (int t = 0; t < 4; ++t) {
      const int n = n0 + t * 16 + (lane & 15);
      v8f c = acc[hlf][t];
      if (MODE == 0) {
        const float bv = bias[n];
#pragma unroll
        for (int r = 0; r < 8; ++r)
          Yf[(size_t)(mb + r) * N + n] = c[r] + bv;
      } else if (MODE == 1) {
#pragma unroll
        for (int r = 0; r < 8; ++r)
          Yb[(size_t)(mb + r) * N + n] = (__bf16)c[r];
      } else {  // MODE 2: Vt[b,h,d,s]; strips are head-aligned (64 == d_k)
        const int h = n >> 6, d = n & 63;
#pragma unroll
        for (int r = 0; r < 8; ++r) {
          const int m = mb + r, b = m >> 11, s = m & (SEQ - 1);
          Yb[((size_t)(b * NUM_HEADS + h) * D_K + d) * SEQ + s] = (__bf16)c[r];
        }
      }
    }
  }
}

// ---------------------------------------------------------------------------
// Scores: P[b,h,q,k] = (Q_h @ K_h^T)/8. One wave = 32(q) x 64(k) tile.
// ---------------------------------------------------------------------------
__global__ void mha_scores(const __bf16* __restrict__ Qb, const __bf16* __restrict__ Kb,
                           float* __restrict__ P) {
  const int lane = threadIdx.x & 31;
  const int bh = blockIdx.y, b = bh / NUM_HEADS, h = bh % NUM_HEADS;
  const int w  = blockIdx.x * (blockDim.x >> 5) + (threadIdx.x >> 5); // 0..2047
  const int q0 = (w >> 5) * 32;
  const int k0 = (w & 31) * 64;

  const __bf16* Qh = Qb + (size_t)b * SEQ * D_MODEL + h * D_K;
  const __bf16* Kh = Kb + (size_t)b * SEQ * D_MODEL + h * D_K;

  v8f acc[2][4] = {};
#pragma unroll
  for (int kc = 0; kc < D_K; kc += 32) {
    v16bf a0 = load_a_bf16(Qh, D_MODEL, q0,      kc, lane);
    v16bf a1 = load_a_bf16(Qh, D_MODEL, q0 + 16, kc, lane);
#pragma unroll
    for (int t = 0; t < 4; ++t) {
      v16bf bt = load_bT_bf16(Kh, D_MODEL, kc, k0 + t * 16, lane);
      acc[0][t] = wmma_bf16(a0, bt, acc[0][t]);
      acc[1][t] = wmma_bf16(a1, bt, acc[1][t]);
    }
  }

  float* Pbh = P + (size_t)bh * SEQ * SEQ;
#pragma unroll
  for (int hlf = 0; hlf < 2; ++hlf) {
    const int mb = q0 + hlf * 16 + ((lane >> 4) << 3);
#pragma unroll
    for (int t = 0; t < 4; ++t) {
      const int n = k0 + t * 16 + (lane & 15);
      v8f c = acc[hlf][t];
#pragma unroll
      for (int r = 0; r < 8; ++r)
        Pbh[(size_t)(mb + r) * SEQ + n] = c[r] * 0.125f;
    }
  }
}

// ---------------------------------------------------------------------------
// Row softmax, in place (f32, required output) + bf16 copy for the PV GEMM.
// Whole 2048-row in registers (8/thread), wave32 reductions.
// ---------------------------------------------------------------------------
__device__ __forceinline__ float block_max256(float x) {
  __shared__ float red[8];
#pragma unroll
  for (int o = 16; o > 0; o >>= 1) x = fmaxf(x, __shfl_xor(x, o, 32));
  if ((threadIdx.x & 31) == 0) red[threadIdx.x >> 5] = x;
  __syncthreads();
  float r = red[0];
#pragma unroll
  for (int i = 1; i < 8; ++i) r = fmaxf(r, red[i]);
  __syncthreads();
  return r;
}

__device__ __forceinline__ float block_sum256(float x) {
  __shared__ float red[8];
#pragma unroll
  for (int o = 16; o > 0; o >>= 1) x += __shfl_xor(x, o, 32);
  if ((threadIdx.x & 31) == 0) red[threadIdx.x >> 5] = x;
  __syncthreads();
  float r = red[0];
#pragma unroll
  for (int i = 1; i < 8; ++i) r += red[i];
  __syncthreads();
  return r;
}

__global__ void mha_softmax(float* __restrict__ P, __bf16* __restrict__ Pb) {
  float*  p  = P  + (size_t)blockIdx.x * SEQ;
  __bf16* pb = Pb + (size_t)blockIdx.x * SEQ;
  const int t = threadIdx.x;
  float v[8];
#pragma unroll
  for (int i = 0; i < 8; ++i) v[i] = p[t + i * 256];
  float m = v[0];
#pragma unroll
  for (int i = 1; i < 8; ++i) m = fmaxf(m, v[i]);
  m = block_max256(m);
  float s = 0.0f;
#pragma unroll
  for (int i = 0; i < 8; ++i) { v[i] = __expf(v[i] - m); s += v[i]; }
  s = block_sum256(s);
  const float inv = 1.0f / s;
#pragma unroll
  for (int i = 0; i < 8; ++i) {
    const float y = v[i] * inv;
    p [t + i * 256] = y;
    pb[t + i * 256] = (__bf16)y;
  }
}

// ---------------------------------------------------------------------------
// PV: O_h = P @ V_h, written into [B,S,768] bf16 at column h*64.
// A = Pbf (bf16 row-major), B = Vt[b,h,d,s] (same two-chunk k layout as A).
// One wave = 32(q) x 64(d) tile; 64 k-steps.
// ---------------------------------------------------------------------------
__global__ void mha_pv(const __bf16* __restrict__ Pb, const __bf16* __restrict__ Vt,
                       __bf16* __restrict__ Ob) {
  const int lane = threadIdx.x & 31;
  const int bh = blockIdx.y, b = bh / NUM_HEADS, h = bh % NUM_HEADS;
  const int q0 = (blockIdx.x * (blockDim.x >> 5) + (threadIdx.x >> 5)) * 32;

  const __bf16* Pbh = Pb + (size_t)bh * SEQ * SEQ;
  const __bf16* Vh  = Vt + (size_t)bh * D_K * SEQ;   // row d: Vh[d*SEQ + s]

  v8f acc[2][4] = {};
  for (int kc = 0; kc < SEQ; kc += 32) {
    if (kc + 32 < SEQ)
      __builtin_prefetch(Pbh + (size_t)(q0 + lane) * SEQ + kc + 32, 0, 1);
    v16bf a0 = load_a_bf16(Pbh, SEQ, q0,      kc, lane);
    v16bf a1 = load_a_bf16(Pbh, SEQ, q0 + 16, kc, lane);
#pragma unroll
    for (int t = 0; t < 4; ++t) {
      v16bf bt = load_bT_bf16(Vh, SEQ, kc, t * 16, lane);
      acc[0][t] = wmma_bf16(a0, bt, acc[0][t]);
      acc[1][t] = wmma_bf16(a1, bt, acc[1][t]);
    }
  }

  __bf16* Oh = Ob + (size_t)b * SEQ * D_MODEL + h * D_K;
#pragma unroll
  for (int hlf = 0; hlf < 2; ++hlf) {
    const int mb = q0 + hlf * 16 + ((lane >> 4) << 3);
#pragma unroll
    for (int t = 0; t < 4; ++t) {
      const int n = t * 16 + (lane & 15);
      v8f c = acc[hlf][t];
#pragma unroll
      for (int r = 0; r < 8; ++r)
        Oh[(size_t)(mb + r) * D_MODEL + n] = (__bf16)c[r];
    }
  }
}

// ---------------------------------------------------------------------------
// Host orchestration (graph-capture safe: launches only, all on `stream`).
// ---------------------------------------------------------------------------
extern "C" void kernel_launch(void* const* d_in, const int* in_sizes, int n_in,
                              void* d_out, int out_size, void* d_ws, size_t ws_size,
                              hipStream_t stream) {
  (void)in_sizes; (void)n_in; (void)out_size; (void)ws_size;
  const float* q  = (const float*)d_in[0];
  const float* k  = (const float*)d_in[1];
  const float* v  = (const float*)d_in[2];
  const float* Wq = (const float*)d_in[3];
  const float* Wk = (const float*)d_in[4];
  const float* Wv = (const float*)d_in[5];
  const float* Wo = (const float*)d_in[6];
  const float* bo = (const float*)d_in[7];

  const size_t SZ  = (size_t)BATCH * SEQ * D_MODEL;     // 3,145,728
  const size_t WSZ = (size_t)D_MODEL * D_MODEL;         // 589,824
  float* out   = (float*)d_out;                         // [B,S,D]
  float* attnW = out + SZ;                              // [B,H,S,S]

  __bf16* wsb = (__bf16*)d_ws;   // ~231 MB total bf16 scratch
  __bf16* Qbf = wsb;
  __bf16* Kbf = Qbf + SZ;
  __bf16* Vt  = Kbf + SZ;        // B*H*64*2048 == SZ elements
  __bf16* Obf = Vt  + SZ;
  __bf16* WtQ = Obf + SZ;
  __bf16* WtK = WtQ + WSZ;
  __bf16* WtV = WtK + WSZ;
  __bf16* WtO = WtV + WSZ;
  __bf16* Pbf = WtO + WSZ;       // B*H*S*S bf16 = 201 MB

  const int M = BATCH * SEQ;     // 4096

  // Weight transpose+convert: 768*768/256 = 2304 blocks each.
  mha_transpose_convert<<<2304, 256, 0, stream>>>(Wq, WtQ);
  mha_transpose_convert<<<2304, 256, 0, stream>>>(Wk, WtK);
  mha_transpose_convert<<<2304, 256, 0, stream>>>(Wv, WtV);
  mha_transpose_convert<<<2304, 256, 0, stream>>>(Wo, WtO);

  // Projections: (4096/32)*(768/64) = 1536 waves -> 192 blocks of 8 waves.
  mha_gemm<1, false><<<192, 256, 0, stream>>>(q, nullptr, WtQ, nullptr,
                                              nullptr, Qbf, M, D_MODEL, D_MODEL);
  mha_gemm<1, false><<<192, 256, 0, stream>>>(k, nullptr, WtK, nullptr,
                                              nullptr, Kbf, M, D_MODEL, D_MODEL);
  mha_gemm<2, false><<<192, 256, 0, stream>>>(v, nullptr, WtV, nullptr,
                                              nullptr, Vt, M, D_MODEL, D_MODEL);

  // Scores: per (b,h) (2048/32)*(2048/64) = 2048 waves -> 256 blocks.
  mha_scores<<<dim3(256, BATCH * NUM_HEADS), 256, 0, stream>>>(Qbf, Kbf, attnW);

  // Softmax: one block per row.
  mha_softmax<<<BATCH * NUM_HEADS * SEQ, 256, 0, stream>>>(attnW, Pbf);

  // PV: per (b,h) 2048/32 = 64 waves -> 8 blocks.
  mha_pv<<<dim3(8, BATCH * NUM_HEADS), 256, 0, stream>>>(Pbf, Vt, Obf);

  // Output projection with bias (bf16 A operand), f32 result.
  mha_gemm<0, true><<<192, 256, 0, stream>>>(nullptr, Obf, WtO, bo,
                                             out, nullptr, M, D_MODEL, D_MODEL);
}
